// OriLinearGNN_38560216383547
// MI455X (gfx1250) — matrix operand order repaired and checked by reading, not compiled, not run on hardware
//
#include <hip/hip_runtime.h>
#include <hip/hip_bf16.h>

typedef __attribute__((ext_vector_type(16))) _Float16 v16h;
typedef __attribute__((ext_vector_type(8)))  float    v8f;

#define LN        128
#define S_DIM     16
#define T_STEPS   4
#define MU_OVER_S 0.05625f   // 0.9 / 16

union Frag { uint4 q[2]; v16h v; };

// ---------------- preprocessing ----------------

// f32 feature table -> f16 (halves gather bandwidth for the edge GEMM)
__global__ void k_cvt_feat(const float* __restrict__ f32, _Float16* __restrict__ f16, int n) {
    int i = blockIdx.x * 256 + threadIdx.x;
    if (i < n) f16[i] = (_Float16)f32[i];
}

// xi_W (256 x 256, row-major in->out) -> WMMA B-fragment order, f16.
// out[((kt*16+nt)*32+lane)*16 + h] = xi_W[K*256 + n],  K = kt*32 + 8*(lane/16) + (h&7) + 16*(h/8),
// n = nt*16 + lane%16  (per ISA 16-bit B 32x16 layout)
__global__ void k_prep_xi(const float* __restrict__ xiW, _Float16* __restrict__ out) {
    int o = blockIdx.x * 256 + threadIdx.x;        // 65536 total
    int h    = o & 15;
    int lane = (o >> 4) & 31;
    int t    = o >> 9;
    int nt = t & 15, kt = t >> 4;
    int K = kt * 32 + ((lane >> 4) << 3) + (h & 7) + ((h >> 3) << 4);
    int n = nt * 16 + (lane & 15);
    out[o] = (_Float16)xiW[K * 256 + n];
}

// rou_W (128 x 16) -> fragment order (4 K-tiles, single N-tile)
__global__ void k_prep_rou(const float* __restrict__ rouW, _Float16* __restrict__ out) {
    int o = blockIdx.x * 256 + threadIdx.x;        // 2048 total
    int h    = o & 15;
    int lane = (o >> 4) & 31;
    int kt   = o >> 9;
    int K = kt * 32 + ((lane >> 4) << 3) + (h & 7) + ((h >> 3) << 4);
    int n = lane & 15;
    out[o] = (_Float16)rouW[K * 16 + n];
}

// ---------------- edge GEMM ----------------
// 64 edges (4 M-tiles) per workgroup: each B fragment fetched from L2 feeds
// 4 WMMAs (4x less L2 traffic than 16-edge blocks).
// A = tanh(X @ xi_W + xi_b) * (mu/s)/dg ; b = tanh(f_node @ rou_W + rou_b)

__global__ __launch_bounds__(256) void k_edge_gemm(
    const int* __restrict__ Xn, const int* __restrict__ Xm, const int* __restrict__ dg,
    const _Float16* __restrict__ feat16,
    const _Float16* __restrict__ xiB16, const _Float16* __restrict__ rouB16,
    const float* __restrict__ xib, const float* __restrict__ roub,
    _Float16* __restrict__ A16, float* __restrict__ bvec, int E)
{
    __shared__ uint4 sXq[64 * 32];   // 64 edges x 256 f16 concat features (32 KB)
    __shared__ float sScale[64];     // (mu/s)/dg per edge

    const int tid = threadIdx.x;
    const int e0  = blockIdx.x * 64;

    if (tid < 64) {
        int e = e0 + tid; if (e >= E) e = E - 1;
        sScale[tid] = MU_OVER_S / (float)dg[e];
    }

    // Stage concat(f[src], f[dst]) for 64 edges, 16B chunks (clamped for tail block).
    for (int idx = tid; idx < 2048; idx += 256) {
        int row = idx >> 5;
        int c   = idx & 31;
        int e   = e0 + row; if (e >= E) e = E - 1;
        const _Float16* src = (c < 16)
            ? (feat16 + (size_t)Xn[e] * LN + c * 8)
            : (feat16 + (size_t)Xm[e] * LN + (c - 16) * 8);
        sXq[row * 32 + c] = *(const uint4*)src;
    }
    __syncthreads();

    const int wave = tid >> 5;        // 8 waves of 32 (wave32)
    const int lane = tid & 31;
    const int hi   = lane >> 4;
    const int lm   = lane & 15;
    const uint32_t* sXd = (const uint32_t*)sXq;

    // Each wave: N-tiles {wave, wave+8} across 4 M-tiles. N-tile nt == row s of A[e].
    const int nt0 = wave, nt1 = wave + 8;
    v8f acc[4][2] = {};

    for (int kt = 0; kt < 8; ++kt) {
        Frag b0, b1;
        const uint4* bp0 = (const uint4*)(xiB16 + ((size_t)((kt * 16 + nt0) * 32 + lane) << 4));
        const uint4* bp1 = (const uint4*)(xiB16 + ((size_t)((kt * 16 + nt1) * 32 + lane) << 4));
        b0.q[0] = bp0[0]; b0.q[1] = bp0[1];
        b1.q[0] = bp1[0]; b1.q[1] = bp1[1];
        #pragma unroll
        for (int m = 0; m < 4; ++m) {
            Frag a;
            int d0 = (m * 16 + lm) * 128 + kt * 16 + 4 * hi;  // A-frag: K = 8*hi + h + 8*(h>=8)
            a.q[0] = *(const uint4*)(sXd + d0);
            a.q[1] = *(const uint4*)(sXd + d0 + 8);
            acc[m][0] = __builtin_amdgcn_wmma_f32_16x16x32_f16(false, a.v, false, b0.v, (short)0, acc[m][0], false, false);
            acc[m][1] = __builtin_amdgcn_wmma_f32_16x16x32_f16(false, a.v, false, b1.v, (short)0, acc[m][1], false, false);
        }
    }

    // Epilogue: D elem (M = r + 8*hi, t = lm); store A[e][s=nt][t] as f16.
    {
        float bias0 = xib[nt0 * 16 + lm];
        float bias1 = xib[nt1 * 16 + lm];
        #pragma unroll
        for (int m = 0; m < 4; ++m) {
            if (e0 + m * 16 < E) {                    // whole M-tile in range (E % 16 == 0)
                #pragma unroll
                for (int r = 0; r < 8; ++r) {
                    int mr = m * 16 + r + 8 * hi;
                    float sc = sScale[mr];
                    size_t base = (size_t)(e0 + mr) * 256;
                    A16[base + nt0 * 16 + lm] = (_Float16)(tanhf(acc[m][0][r] + bias0) * sc);
                    A16[base + nt1 * 16 + lm] = (_Float16)(tanhf(acc[m][1][r] + bias1) * sc);
                }
            }
        }
    }

    // rou tile (K = 128 over node-embed half): waves 0..3, one M-tile each
    // (wave-uniform branches -> EXEC all-ones for WMMA).
    if (wave < 4) {
        const int m = wave;
        if (e0 + m * 16 < E) {
            v8f accr = {};
            for (int kt = 0; kt < 4; ++kt) {
                Frag a;
                int d0 = (m * 16 + lm) * 128 + kt * 16 + 4 * hi;
                a.q[0] = *(const uint4*)(sXd + d0);
                a.q[1] = *(const uint4*)(sXd + d0 + 8);
                Frag b;
                const uint4* bp = (const uint4*)(rouB16 + ((size_t)(kt * 32 + lane) << 4));
                b.q[0] = bp[0]; b.q[1] = bp[1];
                accr = __builtin_amdgcn_wmma_f32_16x16x32_f16(false, a.v, false, b.v, (short)0, accr, false, false);
            }
            float bias = roub[lm];
            #pragma unroll
            for (int r = 0; r < 8; ++r) {
                int e = e0 + m * 16 + r + 8 * hi;
                bvec[(size_t)e * 16 + lm] = tanhf(accr[r] + bias);
            }
        }
    }
}

// ---------------- message passing ----------------

__global__ void k_zero(float* __restrict__ p, int n) {
    int i = blockIdx.x * 256 + threadIdx.x;
    if (i < n) p[i] = 0.0f;
}

// H[e] = A[e] @ states[X_Node[e]] + b[e]; scatter-add into dst[X_Node[e]].
__global__ __launch_bounds__(256) void k_step(
    const int* __restrict__ Xn, const _Float16* __restrict__ A16,
    const float* __restrict__ bvec, const float* __restrict__ src,
    float* __restrict__ dst, int E)
{
    int gid = blockIdx.x * 256 + threadIdx.x;
    if (gid >= E * 16) return;
    int e = gid >> 4, s = gid & 15;
    int ce = Xn[e];
    float acc = bvec[(size_t)e * 16 + s];
    union { uint4 q[2]; _Float16 h[16]; } A;
    const uint4* ap = (const uint4*)(A16 + (size_t)e * 256 + s * 16);
    A.q[0] = ap[0]; A.q[1] = ap[1];
    const float* hsrc = src + (size_t)ce * 16;
    #pragma unroll
    for (int t = 0; t < 16; ++t) acc += (float)A.h[t] * hsrc[t];
    atomicAdd(dst + (size_t)ce * 16 + s, acc);
}

// ---------------- readout + softmax over nodes ----------------

__global__ void k_logits(const float* __restrict__ feats, const float* __restrict__ states,
                         const float* __restrict__ linW, const float* __restrict__ linb,
                         float* __restrict__ logits, int V)
{
    int v = blockIdx.x * 256 + threadIdx.x;
    if (v >= V) return;
    float a0 = linb[0], a1 = linb[1], a2 = linb[2];
    const float* f = feats + (size_t)v * LN;
    #pragma unroll 4
    for (int k = 0; k < LN; ++k) {
        float x = f[k];
        a0 += x * linW[k * 3 + 0];
        a1 += x * linW[k * 3 + 1];
        a2 += x * linW[k * 3 + 2];
    }
    const float* st = states + (size_t)v * 16;
    #pragma unroll
    for (int k = 0; k < 16; ++k) {
        float x = st[k];
        a0 += x * linW[(LN + k) * 3 + 0];
        a1 += x * linW[(LN + k) * 3 + 1];
        a2 += x * linW[(LN + k) * 3 + 2];
    }
    logits[(size_t)v * 3 + 0] = a0;
    logits[(size_t)v * 3 + 1] = a1;
    logits[(size_t)v * 3 + 2] = a2;
}

__global__ void k_colreduce(const float* __restrict__ logits, float* __restrict__ red, int V) {
    int c = blockIdx.x;            // 3 columns
    int tid = threadIdx.x;
    __shared__ float sm[256];
    float m = -3.4e38f;
    for (int v = tid; v < V; v += 256) m = fmaxf(m, logits[(size_t)v * 3 + c]);
    sm[tid] = m; __syncthreads();
    for (int s = 128; s > 0; s >>= 1) { if (tid < s) sm[tid] = fmaxf(sm[tid], sm[tid + s]); __syncthreads(); }
    float mx = sm[0]; __syncthreads();
    float sum = 0.0f;
    for (int v = tid; v < V; v += 256) sum += expf(logits[(size_t)v * 3 + c] - mx);
    sm[tid] = sum; __syncthreads();
    for (int s = 128; s > 0; s >>= 1) { if (tid < s) sm[tid] += sm[tid + s]; __syncthreads(); }
    if (tid == 0) { red[c * 2] = mx; red[c * 2 + 1] = sm[0]; }
}

__global__ void k_softmax(const float* __restrict__ logits, const float* __restrict__ red,
                          float* __restrict__ out, int V)
{
    int i = blockIdx.x * 256 + threadIdx.x;
    if (i >= V * 3) return;
    int c = i % 3;
    out[i] = expf(logits[i] - red[c * 2]) / red[c * 2 + 1];
}

// ---------------- launcher ----------------

extern "C" void kernel_launch(void* const* d_in, const int* in_sizes, int n_in,
                              void* d_out, int out_size, void* d_ws, size_t ws_size,
                              hipStream_t stream)
{
    (void)n_in; (void)out_size; (void)ws_size;
    const int*   Xn    = (const int*)d_in[0];
    const int*   Xm    = (const int*)d_in[1];
    const int*   dg    = (const int*)d_in[2];
    const float* feats = (const float*)d_in[3];
    const float* xiW   = (const float*)d_in[4];
    const float* xib   = (const float*)d_in[5];
    const float* rouW  = (const float*)d_in[6];
    const float* roub  = (const float*)d_in[7];
    const float* linW  = (const float*)d_in[8];
    const float* linb  = (const float*)d_in[9];

    const int E = in_sizes[0];
    const int V = in_sizes[3] / LN;

    char* base = (char*)d_ws;
    size_t off = 0;
    auto alloc = [&](size_t bytes) -> void* {
        void* p = base + off;
        off = (off + bytes + 255) & ~(size_t)255;
        return p;
    };
    _Float16* feat16 = (_Float16*)alloc((size_t)V * LN * 2);
    _Float16* xiB16  = (_Float16*)alloc((size_t)65536 * 2);
    _Float16* rouB16 = (_Float16*)alloc((size_t)2048 * 2);
    _Float16* A16    = (_Float16*)alloc((size_t)E * 256 * 2);
    float*    bvec   = (float*)alloc((size_t)E * 16 * 4);
    float*    s0     = (float*)alloc((size_t)V * 16 * 4);
    float*    s1     = (float*)alloc((size_t)V * 16 * 4);
    float*    logit  = (float*)alloc((size_t)V * 3 * 4);
    float*    red    = (float*)alloc(6 * 4);

    int nf = V * LN;
    k_cvt_feat<<<(nf + 255) / 256, 256, 0, stream>>>(feats, feat16, nf);
    k_prep_xi <<<65536 / 256, 256, 0, stream>>>(xiW, xiB16);
    k_prep_rou<<<2048 / 256, 256, 0, stream>>>(rouW, rouB16);

    k_edge_gemm<<<(E + 63) / 64, 256, 0, stream>>>(Xn, Xm, dg, feat16, xiB16, rouB16,
                                                   xib, roub, A16, bvec, E);

    k_zero<<<(V * 16 + 255) / 256, 256, 0, stream>>>(s0, V * 16);
    float* src = s0; float* dst = s1;
    for (int step = 0; step < T_STEPS; ++step) {
        k_zero<<<(V * 16 + 255) / 256, 256, 0, stream>>>(dst, V * 16);
        k_step<<<(E * 16 + 255) / 256, 256, 0, stream>>>(Xn, A16, bvec, src, dst, E);
        float* t = src; src = dst; dst = t;
    }

    k_logits   <<<(V + 255) / 256, 256, 0, stream>>>(feats, src, linW, linb, logit, V);
    k_colreduce<<<3, 256, 0, stream>>>(logit, red, V);
    k_softmax  <<<(V * 3 + 255) / 256, 256, 0, stream>>>(logit, red, (float*)d_out, V);
}
